// MultiHeadedAttentionCrossAtt_23648089932131
// MI455X (gfx1250) — matrix operand order repaired and checked by reading, not compiled, not run on hardware
//
#include <hip/hip_runtime.h>
#include <stdint.h>

// ---------------------------------------------------------------------------
// CDNA5 (gfx1250) cross-attention, bf16 WMMA + fp32 accumulate.
// B=4, T1=1024, T2=2048, F=1024, H=16, DK=64.  All strides compile-time.
// Workspace (56 MB):
//   [ 0.. 8MB) q_ws bf16 [BH][T1][64]      (pre-scaled by 1/8)
//   [ 8..24MB) ktp  bf16 [BH][32][T2][2]   (K^T, dk pair-interleaved -> uint)
//   [24..40MB) vp   bf16 [BH][T2/2][64][2] (V, t2 pair-interleaved -> uint)
//   [40..56MB) o_ws f32  [B*T1][F]
// ---------------------------------------------------------------------------

typedef __bf16 bf16_t;
typedef __bf16   v16bf __attribute__((ext_vector_type(16)));
typedef float    v8f   __attribute__((ext_vector_type(8)));
typedef uint32_t v8ui  __attribute__((ext_vector_type(8)));

__device__ __forceinline__ uint16_t bfbits(float f) {
  uint32_t u = __builtin_bit_cast(uint32_t, f);
  return (uint16_t)((u + 0x7FFFu + ((u >> 16) & 1u)) >> 16);  // RNE
}
__device__ __forceinline__ bf16_t f2bf(float f) {
  return __builtin_bit_cast(bf16_t, bfbits(f));
}
__device__ __forceinline__ uint32_t pack2(float lo, float hi) {
  return (uint32_t)bfbits(lo) | ((uint32_t)bfbits(hi) << 16);
}

__device__ __forceinline__ v8f wmma_bf16(v8ui a, v8ui b, v8f c) {
  return __builtin_amdgcn_wmma_f32_16x16x32_bf16(
      false, __builtin_bit_cast(v16bf, a),
      false, __builtin_bit_cast(v16bf, b),
      (short)0, c, false, false);
}

// A-fragment (16x32 bf16): lane row = lane&15; VGPR i holds K-pair
//   kpair(i) = (i<4 ? i : 8+(i-4)) + 4*half  (in uint32/pair units)
// -> two aligned uint4 loads per fragment when row stride % 16B == 0.
__device__ __forceinline__ v8ui load_afrag_u(const uint32_t* rowp, int off) {
  uint4 lo = *(const uint4*)(rowp + off);
  uint4 hi = *(const uint4*)(rowp + off + 8);
  v8ui u;
  u[0] = lo.x; u[1] = lo.y; u[2] = lo.z; u[3] = lo.w;
  u[4] = hi.x; u[5] = hi.y; u[6] = hi.z; u[7] = hi.w;
  return u;
}
// B-fragment (32x16 bf16): lane col = lane&15; VGPR i holds K-pair index
//   kpair(i) = (i<4 ? i : 8+(i-4)) + 4*half  (same as A).

// ---------------------------------------------------------------------------
// Projection GEMM: C[M,N] = A[M,1024](f32) @ W[1024,N](f32) + bias
// MODE 0: Q -> bf16 [B,16,T,64] *0.125 | MODE 1: KV split | MODE 2: f32
// ---------------------------------------------------------------------------
template <int MODE, int NCOLS, int TLOG>
__global__ __launch_bounds__(256)
void gemm_proj(const float* __restrict__ A, const float* __restrict__ W,
               const float* __restrict__ bias,
               bf16_t* __restrict__ out_q, bf16_t* __restrict__ out_ktp,
               bf16_t* __restrict__ out_vp, float* __restrict__ out_f) {
  constexpr int K = 1024;
  constexpr int TDIM = 1 << TLOG;

  __shared__ uint32_t AsP[128][36];   // 128 rows x 32 K-pairs (+4 pad) = 144B/row
  __shared__ uint32_t WsP[32][132];   // 32 K-pairs x 128 cols (+4 pad)

  const int tid  = threadIdx.x;
  const int lane = tid & 31;
  const int wave = tid >> 5;
  const int half = lane >> 4;
  const int ln   = lane & 15;

  const int bm = blockIdx.x * 128;
  const int bn = blockIdx.y * 128;
  const int wm = (wave >> 2) * 64;
  const int wn = (wave & 3) * 32;

  v8f acc[4][2];
#pragma unroll
  for (int mt = 0; mt < 4; ++mt)
#pragma unroll
    for (int nt = 0; nt < 2; ++nt)
      acc[mt][nt] = v8f{0.f, 0.f, 0.f, 0.f, 0.f, 0.f, 0.f, 0.f};

  for (int k0 = 0; k0 < K; k0 += 64) {
    // ---- stage A tile (128x64 f32 -> packed bf16 pairs) ----
#pragma unroll
    for (int it = 0; it < 8; ++it) {
      int f4i = tid + it * 256;              // 2048 float4
      int r   = f4i >> 4;
      int c4  = (f4i & 15) * 4;
      float4 v = *(const float4*)(A + (((size_t)(bm + r)) << 10) + k0 + c4);
      uint2 p;
      p.x = pack2(v.x, v.y);
      p.y = pack2(v.z, v.w);
      *(uint2*)&AsP[r][c4 >> 1] = p;
    }
    // ---- stage W tile (64x128) pair-interleaved: WsP[kpair][col] ----
    {
      int kp = tid >> 3;                     // 0..31
      int n0 = (tid & 7) * 16;               // 16 cols per thread
      const float* r0 = W + (size_t)(k0 + 2 * kp)     * NCOLS + bn + n0;
      const float* r1 = W + (size_t)(k0 + 2 * kp + 1) * NCOLS + bn + n0;
#pragma unroll
      for (int g = 0; g < 4; ++g) {
        float4 a = *(const float4*)(r0 + 4 * g);
        float4 b = *(const float4*)(r1 + 4 * g);
        uint4 u;
        u.x = pack2(a.x, b.x); u.y = pack2(a.y, b.y);
        u.z = pack2(a.z, b.z); u.w = pack2(a.w, b.w);
        *(uint4*)&WsP[kp][n0 + 4 * g] = u;
      }
    }
    __syncthreads();

#pragma unroll
    for (int kc = 0; kc < 2; ++kc) {
      v8ui af[4], bfr[2];
#pragma unroll
      for (int mt = 0; mt < 4; ++mt)
        af[mt] = load_afrag_u(&AsP[wm + mt * 16 + ln][0], kc * 16 + 4 * half);
#pragma unroll
      for (int nt = 0; nt < 2; ++nt) {
        int col = wn + nt * 16 + ln;
#pragma unroll
        for (int i = 0; i < 8; ++i)
          bfr[nt][i] = WsP[kc * 16 + (i < 4 ? i : i + 4) + 4 * half][col];
      }
#pragma unroll
      for (int mt = 0; mt < 4; ++mt)
#pragma unroll
        for (int nt = 0; nt < 2; ++nt)
          acc[mt][nt] = wmma_bf16(af[mt], bfr[nt], acc[mt][nt]);
    }
    __syncthreads();
  }

  // ---- epilogue (all index math = shifts/masks) ----
#pragma unroll
  for (int mt = 0; mt < 4; ++mt) {
#pragma unroll
    for (int nt = 0; nt < 2; ++nt) {
#pragma unroll
      for (int r = 0; r < 8; ++r) {
        int gM = bm + wm + mt * 16 + r + 8 * half;
        int gN = bn + wn + nt * 16 + ln;
        float val = acc[mt][nt][r] + bias[gN];
        int b = gM >> TLOG, t = gM & (TDIM - 1);
        if constexpr (MODE == 0) {
          val *= 0.125f;
          int h = gN >> 6, dk = gN & 63;
          out_q[((((size_t)b << 4) + h) << (TLOG + 6)) + ((size_t)t << 6) + dk] =
              f2bf(val);
        } else if constexpr (MODE == 1) {
          if (gN < 1024) {
            int h = gN >> 6, dk = gN & 63;
            size_t bh = ((size_t)b << 4) + h;
            out_ktp[(((bh << 5) + (dk >> 1)) << (TLOG + 1)) + ((size_t)t << 1) +
                    (dk & 1)] = f2bf(val);
          } else {
            int gn2 = gN - 1024;
            int h = gn2 >> 6, dk = gn2 & 63;
            size_t bh = ((size_t)b << 4) + h;
            out_vp[(((bh << (TLOG - 1)) + (t >> 1)) << 7) + ((size_t)dk << 1) +
                   (t & 1)] = f2bf(val);
          }
        } else {
          out_f[((size_t)gM << 10) + gN] = val;
        }
      }
    }
  }
}

// ---------------------------------------------------------------------------
// Flash attention: 1 wave per 16-row q tile, T2 streamed in chunks of 64.
// ---------------------------------------------------------------------------
#define WPB 4
__global__ __launch_bounds__(WPB * 32)
void attn_kernel(const bf16_t* __restrict__ qv,   // [BH][T1][64]
                 const uint32_t* __restrict__ kt, // [BH][32][T2] pairs
                 const uint32_t* __restrict__ vv, // [BH][T2/2][64] pairs
                 const int*    __restrict__ mask, // [B][T2]
                 float*        __restrict__ o) {  // [B][T1][1024]
  constexpr int T1 = 1024, T2 = 2048;
  __shared__ bf16_t plds[WPB][16][72];   // 144B row stride -> aligned b128 reads

  const int lane = threadIdx.x & 31;
  const int wv   = threadIdx.x >> 5;
  const int half = lane >> 4;
  const int ln   = lane & 15;

  const int gw = blockIdx.x * WPB + wv;
  const int bh = gw >> 6;           // / (T1/16)
  const int qt = gw & 63;
  const int b  = bh >> 4;
  const int h  = bh & 15;

  // Q fragments (dk chunks [0,32) and [32,64)) as packed pairs
  const uint32_t* q32 =
      (const uint32_t*)qv + (((size_t)bh << 10) + qt * 16 + ln) * 32;
  v8ui aq[2];
#pragma unroll
  for (int kc = 0; kc < 2; ++kc) aq[kc] = load_afrag_u(q32, kc * 16 + 4 * half);

  float mrow[8], lrow[8];
  v8f oacc[4];
#pragma unroll
  for (int r = 0; r < 8; ++r) { mrow[r] = -__builtin_inff(); lrow[r] = 0.f; }
#pragma unroll
  for (int nt = 0; nt < 4; ++nt)
    oacc[nt] = v8f{0.f, 0.f, 0.f, 0.f, 0.f, 0.f, 0.f, 0.f};

  const size_t kbase = ((size_t)bh << 5) << 11;        // bh*32*T2
  const size_t vbase = ((size_t)bh << 10) << 6;        // bh*(T2/2)*64
  const int* mptr = mask + ((size_t)b << 11);

  for (int t2c = 0; t2c < T2; t2c += 64) {
    // ---- S = Qs @ K^T, 4 n-subtiles x 2 k-chunks ----
    v8f s[4];
#pragma unroll
    for (int nt = 0; nt < 4; ++nt) {
      s[nt] = v8f{0.f, 0.f, 0.f, 0.f, 0.f, 0.f, 0.f, 0.f};
      int n = t2c + nt * 16 + ln;
#pragma unroll
      for (int kc = 0; kc < 2; ++kc) {
        v8ui bk;
#pragma unroll
        for (int i = 0; i < 8; ++i)
          bk[i] = kt[kbase + ((size_t)(kc * 16 + (i < 4 ? i : i + 4) + 4 * half) << 11) + n];
        s[nt] = wmma_bf16(aq[kc], bk, s[nt]);
      }
    }
    // ---- mask + online softmax ----
    int mk[4];
#pragma unroll
    for (int nt = 0; nt < 4; ++nt) mk[nt] = mptr[t2c + nt * 16 + ln];

#pragma unroll
    for (int r = 0; r < 8; ++r) {
      float sv[4];
#pragma unroll
      for (int nt = 0; nt < 4; ++nt)
        sv[nt] = (mk[nt] != 0) ? s[nt][r] : -__builtin_inff();
      float cm = fmaxf(fmaxf(sv[0], sv[1]), fmaxf(sv[2], sv[3]));
#pragma unroll
      for (int off = 1; off < 16; off <<= 1)
        cm = fmaxf(cm, __shfl_xor(cm, off, 32));
      float mnew = fmaxf(mrow[r], cm);
      bool finite = (mnew > -1e37f);
      float alpha = finite ? __expf(mrow[r] - mnew) : 0.f;
      mrow[r] = mnew;
      float psum = 0.f;
#pragma unroll
      for (int nt = 0; nt < 4; ++nt) {
        float pv = finite ? __expf(sv[nt] - mnew) : 0.f;
        psum += pv;
        plds[wv][r + 8 * half][nt * 16 + ln] = f2bf(pv);
      }
#pragma unroll
      for (int off = 1; off < 16; off <<= 1)
        psum += __shfl_xor(psum, off, 32);
      lrow[r] = lrow[r] * alpha + psum;
#pragma unroll
      for (int nt = 0; nt < 4; ++nt) oacc[nt][r] *= alpha;
    }
    __builtin_amdgcn_wave_barrier();   // fence P writes -> reads (same wave; DS in-order)

    // ---- O += P @ V ----
    v8ui pa[2];
    const uint32_t* prow = (const uint32_t*)&plds[wv][ln][0];
#pragma unroll
    for (int kc = 0; kc < 2; ++kc) pa[kc] = load_afrag_u(prow, kc * 16 + 4 * half);

    const int t2h = t2c >> 1;
#pragma unroll
    for (int nt = 0; nt < 4; ++nt) {
      int col = nt * 16 + ln;
#pragma unroll
      for (int kc = 0; kc < 2; ++kc) {
        v8ui bv;
#pragma unroll
        for (int i = 0; i < 8; ++i)
          bv[i] = vv[vbase + ((size_t)(t2h + kc * 16 + (i < 4 ? i : i + 4) + 4 * half) << 6) + col];
        oacc[nt] = wmma_bf16(pa[kc], bv, oacc[nt]);
      }
    }
    __builtin_amdgcn_wave_barrier();   // protect plds before next overwrite
  }

  // ---- normalize + store ----
#pragma unroll
  for (int r = 0; r < 8; ++r) {
    float inv = (lrow[r] > 0.f) ? 1.f / lrow[r] : 0.f;
    int t1 = qt * 16 + r + 8 * half;
#pragma unroll
    for (int nt = 0; nt < 4; ++nt)
      o[((((size_t)b << 10) + t1) << 10) + (h << 6) + nt * 16 + ln] =
          oacc[nt][r] * inv;
  }
}

// ---------------------------------------------------------------------------
extern "C" void kernel_launch(void* const* d_in, const int* in_sizes, int n_in,
                              void* d_out, int out_size, void* d_ws, size_t ws_size,
                              hipStream_t stream) {
  (void)in_sizes; (void)n_in; (void)out_size; (void)ws_size;
  const float* x   = (const float*)d_in[0];
  const float* mem = (const float*)d_in[1];
  const int*   msk = (const int*)d_in[2];
  const float* Wq  = (const float*)d_in[3];
  const float* bq  = (const float*)d_in[4];
  const float* Wkv = (const float*)d_in[5];
  const float* bkv = (const float*)d_in[6];
  const float* Wo  = (const float*)d_in[7];
  const float* bo  = (const float*)d_in[8];
  float* out = (float*)d_out;

  char* ws = (char*)d_ws;
  bf16_t* q_ws = (bf16_t*)(ws);
  bf16_t* ktp  = (bf16_t*)(ws + (size_t)8  * 1024 * 1024);
  bf16_t* vp   = (bf16_t*)(ws + (size_t)24 * 1024 * 1024);
  float*  o_ws = (float*) (ws + (size_t)40 * 1024 * 1024);

  // Q = x @ Wq + bq (M=4096,N=1024), pre-scaled by DK^-0.5
  gemm_proj<0, 1024, 10><<<dim3(32, 8), 256, 0, stream>>>(
      x, Wq, bq, q_ws, nullptr, nullptr, nullptr);
  // KV = memory @ Wkv + bkv (M=8192,N=2048) -> ktp / vp
  gemm_proj<1, 2048, 11><<<dim3(64, 16), 256, 0, stream>>>(
      mem, Wkv, bkv, nullptr, ktp, vp, nullptr);
  // attention: 4096 q-tiles, 1 wave each
  attn_kernel<<<dim3(4096 / WPB), dim3(WPB * 32), 0, stream>>>(
      q_ws, (const uint32_t*)ktp, (const uint32_t*)vp, msk, o_ws);
  // out = O @ Wo + bo (M=4096,N=1024) f32
  gemm_proj<2, 1024, 10><<<dim3(32, 8), 256, 0, stream>>>(
      o_ws, Wo, bo, nullptr, nullptr, nullptr, out);
}
